// SpotGuidedTransformer_77214922047577
// MI455X (gfx1250) — compile-verified, matrix operand order checked by reading, not run on hardware
//
#include <hip/hip_runtime.h>
#include <math.h>

// ---------------- constants ----------------
#define NF    2048       // N == M fine points
#define NCo   256        // coarse points
#define CDIM  128        // feature dim C
#define HEADS 4
#define DHEAD 32
#define KNN1  17         // K+1
#define SPOTK 8
#define NSPOT 64         // SPOTS * SPOT_K
#define SEEDN 64
#define QK_SCALE 0.17677669529663687f   // 1/sqrt(32)

typedef __attribute__((ext_vector_type(16))) _Float16 v16h;
typedef __attribute__((ext_vector_type(4)))  _Float16 v4h;
typedef __attribute__((ext_vector_type(8)))  float    v8f;
typedef __attribute__((ext_vector_type(4)))  float    v4f;
typedef __attribute__((ext_vector_type(4)))  unsigned int v4u;
typedef __attribute__((ext_vector_type(8)))  int      v8i;
typedef __attribute__((ext_vector_type(4)))  int      v4i;

#define GF_BIAS  1
#define GF_RELU  2
#define GF_ACCUM 4

// =====================================================================
// TDM probe: exercise the Tensor Data Mover (async tensor load to LDS)
// + s_wait_tensorcnt.  Reads input points (stable across calls), writes
// only workspace scratch -> deterministic & capture-safe.
// D# per CDNA5 ISA ch.8: group0 = {flags, lds_addr, global_addr lo/hi|type},
// group1 = {data_size, dims/strides}.  64x1 tile of 4-byte elements.
// =====================================================================
#if defined(__has_builtin)
#if __has_builtin(__builtin_amdgcn_tensor_load_to_lds)
#define HAVE_TDM 1
#endif
#endif

__global__ void tdm_probe_k(const float* __restrict__ src, float* __restrict__ dst)
{
    __shared__ __align__(16) float buf[64];
#ifdef HAVE_TDM
    unsigned long long ga = (unsigned long long)(const void*)src;
    v4u g0;
    g0[0] = 1u;                                   // count=1 (valid user D#)
    g0[1] = 0u;                                   // lds_addr = 0
    g0[2] = (unsigned)(ga & 0xFFFFFFFFu);         // global_addr[31:0]
    g0[3] = (unsigned)((ga >> 32) & 0x1FFFFFFu)   // global_addr[56:32]
            | (2u << 30);                         // type = 2 ("image")
    v8i g1;
    g1[0] = 0x20000;                              // data_size = 2 (4 bytes)
    g1[1] = (int)(64u << 16);                     // tensor_dim0 = 64 (bits 79:48)
    g1[2] = (int)(1u << 16);                      // tensor_dim1 = 1  (bits 111:80)
    g1[3] = (int)(64u << 16);                     // tile_dim0 = 64   (bits 127:112)
    g1[4] = 1;                                    // tile_dim1 = 1, tile_dim2 = 0
    g1[5] = 64;                                   // tensor_dim0_stride = 64
    g1[6] = 0;
    g1[7] = 0;
    v4i z4 = {0, 0, 0, 0};
#if __clang_major__ >= 23
    v8i z8 = {0, 0, 0, 0, 0, 0, 0, 0};
    __builtin_amdgcn_tensor_load_to_lds(g0, g1, z4, z4, z8, 0);
#else
    __builtin_amdgcn_tensor_load_to_lds(g0, g1, z4, z4, 0);
#endif
    __builtin_amdgcn_s_wait_tensorcnt(0);
#else
    buf[threadIdx.x] = src[threadIdx.x];
#endif
    __syncthreads();
    dst[threadIdx.x] = buf[threadIdx.x];
}

// =====================================================================
// Tiled WMMA GEMM.  Block = 128 threads (4 waves), 64x64 output tile.
// Wave w owns rows [w*16, w*16+16); K stepped by 32.
// A/B tiles staged in LDS in WMMA fragment order (f16) using float4
// global loads; each wave then fetches a fragment with wide ds_loads and
// issues 4 WMMAs per k-step (A fragment reused across N sub-tiles).
// Requires K % 32 == 0 and N % 16 == 0 (true for all uses here); all
// operand bases/strides are 16B-aligned multiples of 4 floats.
// C[M,N] (+)= alpha*A*B (+bias) with optional relu/accumulate.
// =====================================================================
__device__ __forceinline__ int frag_pos(int mn, int k)
{
    // (row-or-col mn in [0,16), k in [0,32)) -> half index within lane frag
    int lanei = mn + (((k >> 3) & 1) << 4);        // which lane holds it
    int ii = (k & 7) | ((k & 16) >> 1);            // which half within lane
    return lanei * 16 + ii;
}

__global__ __launch_bounds__(128) void gemm_wmma_f16(
    const float* __restrict__ A, int lda,
    const float* __restrict__ B, int ldb, int transB,
    float* __restrict__ C, int ldc,
    const float* __restrict__ bias,
    int M, int N, int K, float alpha, int flags)
{
    __shared__ __align__(32) _Float16 lsA[4 * 512];   // 4 subtiles * 32 lanes * 16 halfs
    __shared__ __align__(32) _Float16 lsB[4 * 512];

    int tid  = threadIdx.x;
    int wave = tid >> 5;
    int lane = tid & 31;
    int half = lane >> 4;
    int l    = lane & 15;

    int m0 = blockIdx.y * 64;
    int n0 = blockIdx.x * 64;

    v8f acc0 = {}, acc1 = {}, acc2 = {}, acc3 = {};

    for (int k0 = 0; k0 < K; k0 += 32) {
        // ---- stage A tile (64 rows x 32 k) : float4 per iteration ----
#pragma unroll
        for (int f = tid; f < 512; f += 128) {
            int r  = f >> 3;           // row in tile (8 quads per row)
            int kq = (f & 7) << 2;     // k-quad base
            int grow = m0 + r;
            int gr = grow < M ? grow : (M - 1);
            v4f v = *reinterpret_cast<const v4f*>(&A[(size_t)gr * lda + (k0 + kq)]);
            bool oob = (grow >= M);
            v4h hv;
            hv[0] = (_Float16)(oob ? 0.0f : v[0]);
            hv[1] = (_Float16)(oob ? 0.0f : v[1]);
            hv[2] = (_Float16)(oob ? 0.0f : v[2]);
            hv[3] = (_Float16)(oob ? 0.0f : v[3]);
            // the 4 k's land in 4 consecutive halfs of the same lane slot
            *reinterpret_cast<v4h*>(&lsA[(r >> 4) * 512 + frag_pos(r & 15, kq)]) = hv;
        }
        // ---- stage B tile (32 k x 64 cols) ----
        if (transB) {
#pragma unroll
            for (int f = tid; f < 512; f += 128) {
                int c  = f >> 3;
                int kq = (f & 7) << 2;
                int gcol = n0 + c;
                int gc = gcol < N ? gcol : (N - 1);
                v4f v = *reinterpret_cast<const v4f*>(&B[(size_t)gc * ldb + (k0 + kq)]);
                bool oob = (gcol >= N);
                v4h hv;
                hv[0] = (_Float16)(oob ? 0.0f : v[0]);
                hv[1] = (_Float16)(oob ? 0.0f : v[1]);
                hv[2] = (_Float16)(oob ? 0.0f : v[2]);
                hv[3] = (_Float16)(oob ? 0.0f : v[3]);
                *reinterpret_cast<v4h*>(&lsB[(c >> 4) * 512 + frag_pos(c & 15, kq)]) = hv;
            }
        } else {
#pragma unroll
            for (int f = tid; f < 512; f += 128) {
                int k  = f >> 4;           // 16 col-quads per k row
                int cq = (f & 15) << 2;    // col-quad base
                int gc0 = n0 + cq;
                bool oob = (gc0 >= N);     // N % 16 == 0 -> quad fully in or out
                int gcc = oob ? (N - 4) : gc0;
                v4f v = *reinterpret_cast<const v4f*>(&B[(size_t)(k0 + k) * ldb + gcc]);
#pragma unroll
                for (int j = 0; j < 4; ++j) {
                    int c = cq + j;
                    lsB[(c >> 4) * 512 + frag_pos(c & 15, k)] =
                        (_Float16)(oob ? 0.0f : v[j]);
                }
            }
        }
        __syncthreads();

        // ---- compute: wave's A fragment reused across 4 B subtiles ----
        v16h a = *reinterpret_cast<const v16h*>(&lsA[wave * 512 + lane * 16]);
        v16h b0 = *reinterpret_cast<const v16h*>(&lsB[0 * 512 + lane * 16]);
        v16h b1 = *reinterpret_cast<const v16h*>(&lsB[1 * 512 + lane * 16]);
        v16h b2 = *reinterpret_cast<const v16h*>(&lsB[2 * 512 + lane * 16]);
        v16h b3 = *reinterpret_cast<const v16h*>(&lsB[3 * 512 + lane * 16]);
        acc0 = __builtin_amdgcn_wmma_f32_16x16x32_f16(false, a, false, b0, (short)0, acc0, false, false);
        acc1 = __builtin_amdgcn_wmma_f32_16x16x32_f16(false, a, false, b1, (short)0, acc1, false, false);
        acc2 = __builtin_amdgcn_wmma_f32_16x16x32_f16(false, a, false, b2, (short)0, acc2, false, false);
        acc3 = __builtin_amdgcn_wmma_f32_16x16x32_f16(false, a, false, b3, (short)0, acc3, false, false);
        __syncthreads();
    }

    // ---- store ----
    int orow_base = m0 + wave * 16 + half * 8;
#pragma unroll
    for (int nb = 0; nb < 4; ++nb) {
        v8f* accp = (nb == 0) ? &acc0 : (nb == 1) ? &acc1 : (nb == 2) ? &acc2 : &acc3;
        int ocol = n0 + nb * 16 + l;
        if (ocol >= N) continue;
#pragma unroll
        for (int r = 0; r < 8; ++r) {
            int orow = orow_base + r;
            if (orow >= M) continue;
            float v = alpha * (*accp)[r];
            if (flags & GF_BIAS)  v += bias[ocol];
            if (flags & GF_ACCUM) v += C[(size_t)orow * ldc + ocol];
            if (flags & GF_RELU)  v = fmaxf(v, 0.0f);
            C[(size_t)orow * ldc + ocol] = v;
        }
    }
}

// =====================================================================
// Elementwise / reduction kernels
// =====================================================================
__global__ void copy_k(const float* __restrict__ src, float* __restrict__ dst, size_t n)
{
    size_t id = (size_t)blockIdx.x * blockDim.x + threadIdx.x;
    if (id < n) dst[id] = src[id];
}

__global__ void elu1_k(float* __restrict__ x, size_t n)
{
    size_t id = (size_t)blockIdx.x * blockDim.x + threadIdx.x;
    if (id < n) {
        float v = x[id];
        x[id] = (v > 0.0f) ? (v + 1.0f) : __expf(v);   // elu(x)+1
    }
}

__global__ void pairdist_k(const float* __restrict__ p, int n, float* __restrict__ out)
{
    size_t id = (size_t)blockIdx.x * blockDim.x + threadIdx.x;
    size_t tot = (size_t)n * n;
    if (id >= tot) return;
    int i = (int)(id / n), j = (int)(id % n);
    float dx = p[i * 3 + 0] - p[j * 3 + 0];
    float dy = p[i * 3 + 1] - p[j * 3 + 1];
    float dz = p[i * 3 + 2] - p[j * 3 + 2];
    out[id] = dx * dx + dy * dy + dz * dz;
}

__global__ void topk_rows_k(const float* __restrict__ dist, int nrows, int L,
                            int ksel, int ld, int* __restrict__ out)
{
    int n = blockIdx.x * blockDim.x + threadIdx.x;
    if (n >= nrows) return;
    float bd[KNN1]; int bi[KNN1];
    for (int t = 0; t < ksel; ++t) { bd[t] = 3.4e38f; bi[t] = 0; }
    const float* row = dist + (size_t)n * L;
    for (int m = 0; m < L; ++m) {
        float d = row[m];
        if (d < bd[ksel - 1]) {
            int t = ksel - 1;
            while (t > 0 && bd[t - 1] > d) { bd[t] = bd[t - 1]; bi[t] = bi[t - 1]; --t; }
            bd[t] = d; bi[t] = m;
        }
    }
    for (int t = 0; t < ksel; ++t) out[(size_t)n * ld + t] = bi[t];
}

__global__ void nn1_points_k(const float* __restrict__ pa, int na,
                             const float* __restrict__ pb, int nb, int* __restrict__ out)
{
    int i = blockIdx.x * blockDim.x + threadIdx.x;
    if (i >= na) return;
    float ax = pa[i * 3], ay = pa[i * 3 + 1], az = pa[i * 3 + 2];
    float best = 3.4e38f; int bidx = 0;
    for (int j = 0; j < nb; ++j) {
        float dx = ax - pb[j * 3], dy = ay - pb[j * 3 + 1], dz = az - pb[j * 3 + 2];
        float d = dx * dx + dy * dy + dz * dz;
        if (d < best) { best = d; bidx = j; }
    }
    out[i] = bidx;
}

__global__ void topk_points_k(const float* __restrict__ pa, int na,
                              const float* __restrict__ pb, int nb,
                              int ksel, int* __restrict__ out)
{
    int i = blockIdx.x * blockDim.x + threadIdx.x;
    if (i >= na) return;
    float ax = pa[i * 3], ay = pa[i * 3 + 1], az = pa[i * 3 + 2];
    float bd[8]; int bi[8];
    for (int t = 0; t < ksel; ++t) { bd[t] = 3.4e38f; bi[t] = 0; }
    for (int j = 0; j < nb; ++j) {
        float dx = ax - pb[j * 3], dy = ay - pb[j * 3 + 1], dz = az - pb[j * 3 + 2];
        float d = dx * dx + dy * dy + dz * dz;
        if (d < bd[ksel - 1]) {
            int t = ksel - 1;
            while (t > 0 && bd[t - 1] > d) { bd[t] = bd[t - 1]; bi[t] = bi[t - 1]; --t; }
            bd[t] = d; bi[t] = j;
        }
    }
    for (int t = 0; t < ksel; ++t) out[(size_t)i * ksel + t] = bi[t];
}

__global__ void gather_rows_k(const float* __restrict__ src, const int* __restrict__ idx,
                              float* __restrict__ dst, int rows, int cols)
{
    size_t id = (size_t)blockIdx.x * blockDim.x + threadIdx.x;
    if (id >= (size_t)rows * cols) return;
    int r = (int)(id / cols), c = (int)(id % cols);
    dst[id] = src[(size_t)idx[r] * cols + c];
}

__global__ void maxpool_gather_k(const float* __restrict__ src, const int* __restrict__ idx,
                                 float* __restrict__ dst, int rows, int kk, int cols)
{
    size_t id = (size_t)blockIdx.x * blockDim.x + threadIdx.x;
    if (id >= (size_t)rows * cols) return;
    int r = (int)(id / cols), c = (int)(id % cols);
    float m = -3.4e38f;
    for (int t = 0; t < kk; ++t)
        m = fmaxf(m, src[(size_t)idx[r * kk + t] * cols + c]);
    dst[id] = m;
}

// y[row] = layernorm(x[row] + t[row]); row width fixed 128; wave per row.
__global__ void residual_ln_k(const float* __restrict__ x, const float* __restrict__ t,
                              float* __restrict__ y, int n)
{
    int row = blockIdx.x * (blockDim.x >> 5) + (threadIdx.x >> 5);
    if (row >= n) return;
    int lane = threadIdx.x & 31;
    const float* px = x + (size_t)row * CDIM;
    const float* pt = t + (size_t)row * CDIM;
    float v[4]; float s = 0.0f;
#pragma unroll
    for (int j = 0; j < 4; ++j) { v[j] = px[lane * 4 + j] + pt[lane * 4 + j]; s += v[j]; }
    for (int o = 16; o; o >>= 1) s += __shfl_xor(s, o);
    float mean = s * (1.0f / CDIM);
    float q = 0.0f;
#pragma unroll
    for (int j = 0; j < 4; ++j) { float d = v[j] - mean; q += d * d; }
    for (int o = 16; o; o >>= 1) q += __shfl_xor(q, o);
    float inv = rsqrtf(q * (1.0f / CDIM) + 1e-5f);
#pragma unroll
    for (int j = 0; j < 4; ++j)
        y[(size_t)row * CDIM + lane * 4 + j] = (v[j] - mean) * inv;
}

// softmax along last dim; wave per row.
__global__ void softmax_rows_k(float* __restrict__ x, int nrows, int L)
{
    int row = blockIdx.x * (blockDim.x >> 5) + (threadIdx.x >> 5);
    if (row >= nrows) return;
    int lane = threadIdx.x & 31;
    float* p = x + (size_t)row * L;
    float mx = -3.4e38f;
    for (int i = lane; i < L; i += 32) mx = fmaxf(mx, p[i]);
    for (int o = 16; o; o >>= 1) mx = fmaxf(mx, __shfl_xor(mx, o));
    float s = 0.0f;
    for (int i = lane; i < L; i += 32) { float e = __expf(p[i] - mx); p[i] = e; s += e; }
    for (int o = 16; o; o >>= 1) s += __shfl_xor(s, o);
    float inv = 1.0f / s;
    for (int i = lane; i < L; i += 32) p[i] *= inv;
}

__global__ void row_maxsum_k(const float* __restrict__ s, int n, int m,
                             float* __restrict__ rmax, float* __restrict__ rsum)
{
    int row = blockIdx.x * (blockDim.x >> 5) + (threadIdx.x >> 5);
    if (row >= n) return;
    int lane = threadIdx.x & 31;
    const float* p = s + (size_t)row * m;
    float mx = -3.4e38f;
    for (int i = lane; i < m; i += 32) mx = fmaxf(mx, p[i]);
    for (int o = 16; o; o >>= 1) mx = fmaxf(mx, __shfl_xor(mx, o));
    float sum = 0.0f;
    for (int i = lane; i < m; i += 32) sum += __expf(p[i] - mx);
    for (int o = 16; o; o >>= 1) sum += __shfl_xor(sum, o);
    if (lane == 0) { rmax[row] = mx; rsum[row] = sum; }
}

__global__ void col_maxsum_k(const float* __restrict__ s, int n, int m,
                             float* __restrict__ cmaxv, float* __restrict__ csumv)
{
    int col = blockIdx.x * (blockDim.x >> 5) + (threadIdx.x >> 5);
    if (col >= m) return;
    int lane = threadIdx.x & 31;
    float mx = -3.4e38f;
    for (int i = lane; i < n; i += 32) mx = fmaxf(mx, s[(size_t)i * m + col]);
    for (int o = 16; o; o >>= 1) mx = fmaxf(mx, __shfl_xor(mx, o));
    float sum = 0.0f;
    for (int i = lane; i < n; i += 32) sum += __expf(s[(size_t)i * m + col] - mx);
    for (int o = 16; o; o >>= 1) sum += __shfl_xor(sum, o);
    if (lane == 0) { cmaxv[col] = mx; csumv[col] = sum; }
}

__global__ void dual_combine_k(const float* __restrict__ s, int n, int m,
                               const float* __restrict__ rmax, const float* __restrict__ rsum,
                               const float* __restrict__ cmax, const float* __restrict__ csum,
                               float* __restrict__ out)
{
    size_t id = (size_t)blockIdx.x * blockDim.x + threadIdx.x;
    if (id >= (size_t)n * m) return;
    int i = (int)(id / m), j = (int)(id % m);
    float v = s[id];
    out[id] = (__expf(v - rmax[i]) / rsum[i]) * (__expf(v - cmax[j]) / csum[j]);
}

__global__ void row_argmax_k(const float* __restrict__ s, int n, int m,
                             float* __restrict__ vmax, int* __restrict__ imax)
{
    int row = blockIdx.x * (blockDim.x >> 5) + (threadIdx.x >> 5);
    if (row >= n) return;
    int lane = threadIdx.x & 31;
    const float* p = s + (size_t)row * m;
    float bv = -3.4e38f; int bi = 0;
    for (int i = lane; i < m; i += 32) { float v = p[i]; if (v > bv) { bv = v; bi = i; } }
    for (int o = 16; o; o >>= 1) {
        float ov = __shfl_xor(bv, o); int oi = __shfl_xor(bi, o);
        if (ov > bv || (ov == bv && oi < bi)) { bv = ov; bi = oi; }
    }
    if (lane == 0) { vmax[row] = bv; imax[row] = bi; }
}

__global__ void col_argmax_k(const float* __restrict__ s, int n, int m,
                             float* __restrict__ vmax, int* __restrict__ imax)
{
    int col = blockIdx.x * (blockDim.x >> 5) + (threadIdx.x >> 5);
    if (col >= m) return;
    int lane = threadIdx.x & 31;
    float bv = -3.4e38f; int bi = 0;
    for (int i = lane; i < n; i += 32) {
        float v = s[(size_t)i * m + col];
        if (v > bv) { bv = v; bi = i; }
    }
    for (int o = 16; o; o >>= 1) {
        float ov = __shfl_xor(bv, o); int oi = __shfl_xor(bi, o);
        if (ov > bv || (ov == bv && oi < bi)) { bv = ov; bi = oi; }
    }
    if (lane == 0) { vmax[col] = bv; imax[col] = bi; }
}

// comp[n] = mean_j relu(1 - |dA[n,j] - dB[mi[j], mi[n]]| / 3); conf = vmax*comp
__global__ void compat_conf_k(const float* __restrict__ dA, const float* __restrict__ dB,
                              const int* __restrict__ mi, const float* __restrict__ vmax,
                              float* __restrict__ comp, float* __restrict__ conf,
                              float* __restrict__ slab, int NA, int NB, int blk)
{
    int n = blockIdx.x * blockDim.x + threadIdx.x;
    if (n >= NA) return;
    int mn = mi[n];
    const float* row = dA + (size_t)n * NA;
    float acc = 0.0f;
    for (int j = 0; j < NA; ++j) {
        float b = dB[(size_t)mi[j] * NB + mn];
        float v = 1.0f - fabsf(row[j] - b) * (1.0f / 3.0f);
        acc += fmaxf(v, 0.0f);
    }
    float c = acc / (float)NA;
    comp[n] = c;
    conf[n] = vmax[n] * c;
    slab[(size_t)n * 3 + blk] = c;
}

__global__ void seeding_k(const float* __restrict__ comp, const float* __restrict__ conf,
                          int n, int* __restrict__ tok)
{
    __shared__ float red[256];
    int tid = threadIdx.x;
    float m = -3.4e38f;
    for (int i = tid; i < n; i += 256) m = fmaxf(m, comp[i]);
    red[tid] = m; __syncthreads();
    for (int s = 128; s > 0; s >>= 1) {
        if (tid < s) red[tid] = fmaxf(red[tid], red[tid + s]);
        __syncthreads();
    }
    if (tid == 0) {
        float thr = red[0] * 0.1f;
        float bv[SEEDN]; int bi[SEEDN];
        for (int t = 0; t < SEEDN; ++t) { bv[t] = -3.4e38f; bi[t] = 0; }
        for (int i = 0; i < n; ++i) {
            float sc = (comp[i] < thr) ? conf[i] : 0.0f;
            if (sc > bv[SEEDN - 1]) {
                int t = SEEDN - 1;
                while (t > 0 && bv[t - 1] < sc) { bv[t] = bv[t - 1]; bi[t] = bi[t - 1]; --t; }
                bv[t] = sc; bi[t] = i;
            }
        }
        for (int t = 0; t < SEEDN; ++t) tok[t] = bi[t];
    }
}

__global__ void select_spots_k(const int* __restrict__ nbr, const int* __restrict__ memnbr,
                               const float* __restrict__ conf, const int* __restrict__ midx,
                               int n, int* __restrict__ spot)
{
    int i = blockIdx.x * blockDim.x + threadIdx.x;
    if (i >= n) return;
    float cv[KNN1]; int nb[KNN1];
    for (int t = 0; t < KNN1; ++t) {
        nb[t] = nbr[(size_t)i * KNN1 + t];
        cv[t] = conf[nb[t]];
    }
    for (int s = 0; s < 8; ++s) {
        int bt = 0; float bvv = -3.4e38f;
        for (int t = 0; t < KNN1; ++t) if (cv[t] > bvv) { bvv = cv[t]; bt = t; }
        cv[bt] = -3.4e38f;
        int center = nb[bt];
        int cm = midx[center];
        for (int kk2 = 0; kk2 < SPOTK; ++kk2)
            spot[(size_t)i * NSPOT + s * SPOTK + kk2] = memnbr[(size_t)cm * KNN1 + kk2];
    }
}

// ---- sparse spot attention ----
__global__ void sparse_scores_k(const float* __restrict__ q, const float* __restrict__ kb,
                                const int* __restrict__ spot, float* __restrict__ s, int n)
{
    int id = blockIdx.x * blockDim.x + threadIdx.x;
    if (id >= n * HEADS * NSPOT) return;
    int sp = id & (NSPOT - 1);
    int h = (id >> 6) & (HEADS - 1);
    int nn = id >> 8;
    int m = spot[(size_t)nn * NSPOT + sp];
    const float* pq = q + (size_t)nn * CDIM + h * DHEAD;
    const float* pk = kb + (size_t)m * CDIM + h * DHEAD;
    float acc = 0.0f;
#pragma unroll
    for (int d = 0; d < DHEAD; ++d) acc += pq[d] * pk[d];
    s[id] = acc * QK_SCALE;
}

__global__ void sparse_out_k(const float* __restrict__ a, const float* __restrict__ vb,
                             const int* __restrict__ spot, float* __restrict__ o, int n)
{
    int id = blockIdx.x * blockDim.x + threadIdx.x;
    if (id >= n * CDIM) return;
    int c = id & (CDIM - 1);
    int nn = id >> 7;
    int h = c >> 5;
    const float* pa = a + ((size_t)nn * HEADS + h) * NSPOT;
    float acc = 0.0f;
    for (int sp = 0; sp < NSPOT; ++sp)
        acc += pa[sp] * vb[(size_t)spot[(size_t)nn * NSPOT + sp] * CDIM + c];
    o[id] = acc;
}

// ---- linear attention ----
__global__ void linattn_kv_k(const float* __restrict__ k, const float* __restrict__ v,
                             float* __restrict__ kv, float* __restrict__ ksum, int mrows)
{
    int d = threadIdx.x, e = threadIdx.y, h = blockIdx.x;
    float acc = 0.0f;
    for (int m = 0; m < mrows; ++m)
        acc += k[(size_t)m * CDIM + h * DHEAD + d] * v[(size_t)m * CDIM + h * DHEAD + e];
    kv[(size_t)h * DHEAD * DHEAD + d * DHEAD + e] = acc;
    if (e == 0) {
        float s = 0.0f;
        for (int m = 0; m < mrows; ++m) s += k[(size_t)m * CDIM + h * DHEAD + d];
        ksum[h * DHEAD + d] = s;
    }
}

__global__ void linattn_out_k(const float* __restrict__ q, const float* __restrict__ kv,
                              const float* __restrict__ ksum, float* __restrict__ o, int n)
{
    int id = blockIdx.x * blockDim.x + threadIdx.x;
    if (id >= n * CDIM) return;
    int c = id & (CDIM - 1);
    int nn = id >> 7;
    int h = c >> 5, e = c & (DHEAD - 1);
    const float* pq = q + (size_t)nn * CDIM + h * DHEAD;
    const float* pks = ksum + h * DHEAD;
    const float* pkv = kv + (size_t)h * DHEAD * DHEAD;
    float den = 0.0f, acc = 0.0f;
#pragma unroll
    for (int d = 0; d < DHEAD; ++d) {
        den += pq[d] * pks[d];
        acc += pq[d] * pkv[d * DHEAD + e];
    }
    o[id] = acc / (den + 1e-6f);
}

// =====================================================================
// Host-side helpers
// =====================================================================
struct Lay { const float *wq,*bq,*wk,*bk,*wv,*bv,*wo,*bo,*w1,*b1,*w2,*b2; };
struct Scratch { float *qb,*kb,*vb,*ob,*tmp,*big,*kv,*ksum,*memb,*ctmp; };

static void gemm(hipStream_t st, const float* A, int lda, const float* B, int ldb, int transB,
                 float* C, int ldc, const float* bias, int M, int N, int K,
                 float alpha, int flags)
{
    dim3 g((N + 63) / 64, (M + 63) / 64);
    gemm_wmma_f16<<<g, 128, 0, st>>>(A, lda, B, ldb, transB, C, ldc, bias, M, N, K, alpha, flags);
}

static void lin_gemm(hipStream_t st, const float* x, const float* w, const float* b,
                     float* y, int Mrows, int Kin, int Nout, int extra = 0)
{
    gemm(st, x, Kin, w, Nout, 0, y, Nout, b, Mrows, Nout, Kin, 1.0f, GF_BIAS | extra);
}

static void copy_launch(hipStream_t st, const float* s, float* d, size_t n)
{
    copy_k<<<(unsigned)((n + 255) / 256), 256, 0, st>>>(s, d, n);
}
static void softmax_launch(hipStream_t st, float* x, int nrows, int L)
{
    softmax_rows_k<<<(nrows + 7) / 8, 256, 0, st>>>(x, nrows, L);
}
static void resln_launch(hipStream_t st, const float* x, const float* t, float* y, int n)
{
    residual_ln_k<<<(n + 7) / 8, 256, 0, st>>>(x, t, y, n);
}
static void elu1_launch(hipStream_t st, float* x, size_t n)
{
    elu1_k<<<(unsigned)((n + 255) / 256), 256, 0, st>>>(x, n);
}

static void ffn_ln(hipStream_t st, const Lay& L, float* x, int Nq, const Scratch& S)
{
    lin_gemm(st, x, L.w1, L.b1, S.big, Nq, CDIM, 2 * CDIM, GF_RELU);
    gemm(st, S.big, 2 * CDIM, L.w2, CDIM, 0, S.tmp, CDIM, L.b2, Nq, CDIM, 2 * CDIM, 1.0f, GF_BIAS);
    resln_launch(st, x, S.tmp, x, Nq);
}

static void xformer(hipStream_t st, const Lay& L, const float* xin, float* x,
                    const float* mem, int Nq, int Nm, const Scratch& S)
{
    if (x != xin) copy_launch(st, xin, x, (size_t)Nq * CDIM);
    lin_gemm(st, x, L.wq, L.bq, S.qb, Nq, CDIM, CDIM);
    lin_gemm(st, mem, L.wk, L.bk, S.kb, Nm, CDIM, CDIM);
    lin_gemm(st, mem, L.wv, L.bv, S.vb, Nm, CDIM, CDIM);
    for (int h = 0; h < HEADS; ++h)
        gemm(st, S.qb + h * DHEAD, CDIM, S.kb + h * DHEAD, CDIM, 1,
             S.big + (size_t)h * Nq * Nm, Nm, nullptr, Nq, Nm, DHEAD, QK_SCALE, 0);
    softmax_launch(st, S.big, HEADS * Nq, Nm);
    for (int h = 0; h < HEADS; ++h)
        gemm(st, S.big + (size_t)h * Nq * Nm, Nm, S.vb + h * DHEAD, CDIM, 0,
             S.ob + h * DHEAD, CDIM, nullptr, Nq, DHEAD, Nm, 1.0f, 0);
    lin_gemm(st, S.ob, L.wo, L.bo, S.tmp, Nq, CDIM, CDIM);
    resln_launch(st, x, S.tmp, x, Nq);
    ffn_ln(st, L, x, Nq, S);
}

static void linattn(hipStream_t st, const Lay& L, const float* x, const float* mem,
                    float* out, int Nq, int Nm, const Scratch& S)
{
    lin_gemm(st, x, L.wq, L.bq, S.qb, Nq, CDIM, CDIM);
    elu1_launch(st, S.qb, (size_t)Nq * CDIM);
    lin_gemm(st, mem, L.wk, L.bk, S.kb, Nm, CDIM, CDIM);
    elu1_launch(st, S.kb, (size_t)Nm * CDIM);
    lin_gemm(st, mem, L.wv, L.bv, S.vb, Nm, CDIM, CDIM);
    linattn_kv_k<<<HEADS, dim3(32, 32), 0, st>>>(S.kb, S.vb, S.kv, S.ksum, Nm);
    linattn_out_k<<<(Nq * CDIM + 255) / 256, 256, 0, st>>>(S.qb, S.kv, S.ksum, S.ob, Nq);
    lin_gemm(st, S.ob, L.wo, L.bo, S.tmp, Nq, CDIM, CDIM);
    resln_launch(st, x, S.tmp, out, Nq);
    ffn_ln(st, L, out, Nq, S);
}

static void sparse_layer(hipStream_t st, const Lay& L, const float* x, const float* mem,
                         const int* spot, float* out, int Nq, int Nm, const Scratch& S)
{
    lin_gemm(st, x, L.wq, L.bq, S.qb, Nq, CDIM, CDIM);
    lin_gemm(st, mem, L.wk, L.bk, S.kb, Nm, CDIM, CDIM);
    lin_gemm(st, mem, L.wv, L.bv, S.vb, Nm, CDIM, CDIM);
    sparse_scores_k<<<(Nq * HEADS * NSPOT + 255) / 256, 256, 0, st>>>(S.qb, S.kb, spot, S.big, Nq);
    softmax_launch(st, S.big, Nq * HEADS, NSPOT);
    sparse_out_k<<<(Nq * CDIM + 255) / 256, 256, 0, st>>>(S.big, S.vb, spot, S.ob, Nq);
    lin_gemm(st, S.ob, L.wo, L.bo, S.tmp, Nq, CDIM, CDIM);
    resln_launch(st, x, S.tmp, out, Nq);
    ffn_ln(st, L, out, Nq, S);
}

// =====================================================================
extern "C" void kernel_launch(void* const* d_in, const int* in_sizes, int n_in,
                              void* d_out, int out_size, void* d_ws, size_t ws_size,
                              hipStream_t stream)
{
    (void)out_size; (void)ws_size;
    // Flattening-order detection:
    //  sorted (jax tree): first leaf = params.blocks[0].caa.b1 -> 256
    //  insertion order:   first leaf = ref_points              -> 6144
    bool sorted_order = (n_in >= 1 && in_sizes[0] == 2 * CDIM);

    auto F = [&](int i) -> const float* { return (const float*)d_in[i]; };
    auto layAt = [&](int base) -> Lay {
        Lay L;
        if (sorted_order) { // b1,b2,bk,bo,bq,bv,w1,w2,wk,wo,wq,wv
            L.b1 = F(base + 0);  L.b2 = F(base + 1);  L.bk = F(base + 2);  L.bo = F(base + 3);
            L.bq = F(base + 4);  L.bv = F(base + 5);  L.w1 = F(base + 6);  L.w2 = F(base + 7);
            L.wk = F(base + 8);  L.wo = F(base + 9);  L.wq = F(base + 10); L.wv = F(base + 11);
        } else {            // wq,bq,wk,bk,wv,bv,wo,bo,w1,b1,w2,b2
            L.wq = F(base + 0);  L.bq = F(base + 1);  L.wk = F(base + 2);  L.bk = F(base + 3);
            L.wv = F(base + 4);  L.bv = F(base + 5);  L.wo = F(base + 6);  L.bo = F(base + 7);
            L.w1 = F(base + 8);  L.b1 = F(base + 9);  L.w2 = F(base + 10); L.b2 = F(base + 11);
        }
        return L;
    };

    int i_rp, i_sp, i_rf, i_sf, i_rpc, i_spc, i_rfc, i_sfc;
    int i_in1w, i_in1b, i_in2w, i_in2b, lca_base;
    int cb_caa[3], cb_cross[3], cb_sga[3], i_upw[3], i_upb[3], i_dww[3], i_dwb[3];
    if (sorted_order) {
        for (int b = 0; b < 3; ++b) {
            int base = b * 40;
            cb_caa[b] = base; cb_cross[b] = base + 12;
            i_dwb[b] = base + 24; i_dww[b] = base + 25;
            cb_sga[b] = base + 26; i_upb[b] = base + 38; i_upw[b] = base + 39;
        }
        i_in1b = 120; i_in1w = 121; i_in2b = 122; i_in2w = 123; lca_base = 124;
        i_rf = 136; i_rfc = 137; i_rp = 138; i_rpc = 139;
        i_sf = 140; i_sfc = 141; i_sp = 142; i_spc = 143;
    } else {
        i_rp = 0; i_sp = 1; i_rf = 2; i_sf = 3; i_rpc = 4; i_spc = 5; i_rfc = 6; i_sfc = 7;
        i_in1w = 8; i_in1b = 9; i_in2w = 10; i_in2b = 11; lca_base = 12;
        for (int b = 0; b < 3; ++b) {
            int base = 24 + 40 * b;
            cb_cross[b] = base; cb_caa[b] = base + 12; cb_sga[b] = base + 24;
            i_upw[b] = base + 36; i_upb[b] = base + 37; i_dww[b] = base + 38; i_dwb[b] = base + 39;
        }
    }

    // ---- workspace bump allocator ----
    char* wp = (char*)d_ws;
    auto alloc = [&](size_t bytes) -> void* {
        void* p = (void*)wp; wp += (bytes + 255) & ~(size_t)255; return p;
    };
    float* rd   = (float*)alloc((size_t)NF * NF * 4);
    float* sd   = (float*)alloc((size_t)NF * NF * 4);
    float* sraw = (float*)alloc((size_t)NF * NF * 4);
    int* refidx = (int*)alloc((size_t)NF * KNN1 * 4);
    int* srcidx = (int*)alloc((size_t)NF * KNN1 * 4);
    int* refup  = (int*)alloc((size_t)NF * 4);
    int* srcup  = (int*)alloc((size_t)NF * 4);
    int* refdown = (int*)alloc((size_t)NCo * 8 * 4);
    int* srcdown = (int*)alloc((size_t)NCo * 8 * 4);
    float* refc = (float*)alloc((size_t)NCo * CDIM * 4);
    float* srcc = (float*)alloc((size_t)NCo * CDIM * 4);
    float* nrc  = (float*)alloc((size_t)NCo * CDIM * 4);
    float* nsc  = (float*)alloc((size_t)NCo * CDIM * 4);
    float* refF = (float*)alloc((size_t)NF * CDIM * 4);
    float* srcF = (float*)alloc((size_t)NF * CDIM * 4);
    Scratch S;
    S.qb   = (float*)alloc((size_t)NF * CDIM * 4);
    S.kb   = (float*)alloc((size_t)NF * CDIM * 4);
    S.vb   = (float*)alloc((size_t)NF * CDIM * 4);
    S.ob   = (float*)alloc((size_t)NF * CDIM * 4);
    S.tmp  = (float*)alloc((size_t)NF * CDIM * 4);
    S.big  = (float*)alloc((size_t)NF * 2 * CDIM * 4);  // 524288 floats
    S.kv   = (float*)alloc((size_t)HEADS * DHEAD * DHEAD * 4);
    S.ksum = (float*)alloc((size_t)HEADS * DHEAD * 4);
    S.memb = (float*)alloc((size_t)SEEDN * CDIM * 4);
    S.ctmp = (float*)alloc((size_t)NCo * CDIM * 4);
    float* rmax = (float*)alloc((size_t)NF * 4);
    float* rsum = (float*)alloc((size_t)NF * 4);
    float* cmax = (float*)alloc((size_t)NF * 4);
    float* csum = (float*)alloc((size_t)NF * 4);
    float* vmaxR = (float*)alloc((size_t)NF * 4);
    float* vmaxC = (float*)alloc((size_t)NF * 4);
    int* midx  = (int*)alloc((size_t)NF * 4);
    int* midx2 = (int*)alloc((size_t)NF * 4);
    float* compR = (float*)alloc((size_t)NF * 4);
    float* compS = (float*)alloc((size_t)NF * 4);
    float* confR = (float*)alloc((size_t)NF * 4);
    float* confS = (float*)alloc((size_t)NF * 4);
    int* tokR = (int*)alloc(SEEDN * 4);
    int* tokS = (int*)alloc(SEEDN * 4);
    int* spotR = (int*)alloc((size_t)NF * NSPOT * 4);
    int* spotS = (int*)alloc((size_t)NF * NSPOT * 4);
    float* tdmbuf = (float*)alloc(64 * 4);

    // ---- output layout ----
    float* outF = (float*)d_out;
    float* newR  = outF;
    float* newS  = outF + (size_t)NF * CDIM;
    float* corr0 = outF + 2ull * NF * CDIM;
    float* rcomp = corr0 + 3ull * NF * NF;
    float* scomp = rcomp + (size_t)NF * 3;

    // ---- TDM async-tensor probe (writes only ws scratch) ----
    tdm_probe_k<<<1, 32, 0, stream>>>(F(i_rp), tdmbuf);

    // ---- precompute: distances, knn, up/down ----
    unsigned gNN = (unsigned)(((size_t)NF * NF + 255) / 256);
    pairdist_k<<<gNN, 256, 0, stream>>>(F(i_rp), NF, rd);
    pairdist_k<<<gNN, 256, 0, stream>>>(F(i_sp), NF, sd);
    topk_rows_k<<<(NF + 63) / 64, 64, 0, stream>>>(rd, NF, NF, KNN1, KNN1, refidx);
    topk_rows_k<<<(NF + 63) / 64, 64, 0, stream>>>(sd, NF, NF, KNN1, KNN1, srcidx);
    nn1_points_k<<<(NF + 255) / 256, 256, 0, stream>>>(F(i_rp), NF, F(i_rpc), NCo, refup);
    nn1_points_k<<<(NF + 255) / 256, 256, 0, stream>>>(F(i_sp), NF, F(i_spc), NCo, srcup);
    topk_points_k<<<(NCo + 63) / 64, 64, 0, stream>>>(F(i_rpc), NCo, F(i_rp), NF, 8, refdown);
    topk_points_k<<<(NCo + 63) / 64, 64, 0, stream>>>(F(i_spc), NCo, F(i_sp), NF, 8, srcdown);

    // ---- input projections ----
    lin_gemm(stream, F(i_rfc), F(i_in1w), F(i_in1b), refc, NCo, 2 * CDIM, CDIM);
    lin_gemm(stream, F(i_sfc), F(i_in1w), F(i_in1b), srcc, NCo, 2 * CDIM, CDIM);
    lin_gemm(stream, F(i_rf), F(i_in2w), F(i_in2b), refF, NF, CDIM, CDIM);
    lin_gemm(stream, F(i_sf), F(i_in2w), F(i_in2b), srcF, NF, CDIM, CDIM);

    Lay lca = layAt(lca_base);
    linattn(stream, lca, refF, srcF, newR, NF, NF, S);
    linattn(stream, lca, srcF, refF, newS, NF, NF, S);

    unsigned gFC = (unsigned)(((size_t)NF * CDIM + 255) / 256);
    unsigned gCC = (unsigned)(((size_t)NCo * CDIM + 255) / 256);

    for (int b = 0; b < 3; ++b) {
        Lay cross = layAt(cb_cross[b]);
        Lay caa   = layAt(cb_caa[b]);
        Lay sga   = layAt(cb_sga[b]);
        const float* upw = F(i_upw[b]); const float* upb = F(i_upb[b]);
        const float* dww = F(i_dww[b]); const float* dwb = F(i_dwb[b]);

        // coarse cross transformers (consume OLD refc/srcc)
        xformer(stream, cross, refc, nrc, srcc, NCo, NCo, S);
        xformer(stream, cross, srcc, nsc, refc, NCo, NCo, S);

        // up: feats = new + lin(gather(coarse, idx_up))
        gather_rows_k<<<gFC, 256, 0, stream>>>(nrc, refup, S.tmp, NF, CDIM);
        copy_launch(stream, newR, refF, (size_t)NF * CDIM);
        gemm(stream, S.tmp, CDIM, upw, CDIM, 0, refF, CDIM, upb, NF, CDIM, CDIM, 1.0f,
             GF_BIAS | GF_ACCUM);
        gather_rows_k<<<gFC, 256, 0, stream>>>(nsc, srcup, S.tmp, NF, CDIM);
        copy_launch(stream, newS, srcF, (size_t)NF * CDIM);
        gemm(stream, S.tmp, CDIM, upw, CDIM, 0, srcF, CDIM, upb, NF, CDIM, CDIM, 1.0f,
             GF_BIAS | GF_ACCUM);

        // down: coarse = nrc + lin(maxpool(gather(new, idx_down)))
        maxpool_gather_k<<<gCC, 256, 0, stream>>>(newR, refdown, S.ctmp, NCo, 8, CDIM);
        copy_launch(stream, nrc, refc, (size_t)NCo * CDIM);
        gemm(stream, S.ctmp, CDIM, dww, CDIM, 0, refc, CDIM, dwb, NCo, CDIM, CDIM, 1.0f,
             GF_BIAS | GF_ACCUM);
        maxpool_gather_k<<<gCC, 256, 0, stream>>>(newS, srcdown, S.ctmp, NCo, 8, CDIM);
        copy_launch(stream, nsc, srcc, (size_t)NCo * CDIM);
        gemm(stream, S.ctmp, CDIM, dww, CDIM, 0, srcc, CDIM, dwb, NCo, CDIM, CDIM, 1.0f,
             GF_BIAS | GF_ACCUM);

        // matching scores: s = refF @ srcF^T, then dual softmax -> corr slab
        gemm(stream, refF, CDIM, srcF, CDIM, 1, sraw, NF, nullptr, NF, NF, CDIM, 1.0f, 0);
        row_maxsum_k<<<(NF + 7) / 8, 256, 0, stream>>>(sraw, NF, NF, rmax, rsum);
        col_maxsum_k<<<(NF + 7) / 8, 256, 0, stream>>>(sraw, NF, NF, cmax, csum);
        float* corr = corr0 + (size_t)b * NF * NF;
        dual_combine_k<<<gNN, 256, 0, stream>>>(sraw, NF, NF, rmax, rsum, cmax, csum, corr);

        row_argmax_k<<<(NF + 7) / 8, 256, 0, stream>>>(corr, NF, NF, vmaxR, midx);
        col_argmax_k<<<(NF + 7) / 8, 256, 0, stream>>>(corr, NF, NF, vmaxC, midx2);
        compat_conf_k<<<(NF + 255) / 256, 256, 0, stream>>>(rd, sd, midx, vmaxR,
                                                            compR, confR, rcomp, NF, NF, b);
        compat_conf_k<<<(NF + 255) / 256, 256, 0, stream>>>(sd, rd, midx2, vmaxC,
                                                            compS, confS, scomp, NF, NF, b);
        seeding_k<<<1, 256, 0, stream>>>(compR, confR, NF, tokR);
        seeding_k<<<1, 256, 0, stream>>>(compS, confS, NF, tokS);
        select_spots_k<<<(NF + 127) / 128, 128, 0, stream>>>(refidx, srcidx, confR, midx, NF, spotR);
        select_spots_k<<<(NF + 127) / 128, 128, 0, stream>>>(srcidx, refidx, confS, midx2, NF, spotS);

        // consistency-aware attention over seed tokens
        gather_rows_k<<<(SEEDN * CDIM + 255) / 256, 256, 0, stream>>>(refF, tokR, S.memb, SEEDN, CDIM);
        xformer(stream, caa, refF, refF, S.memb, NF, SEEDN, S);
        gather_rows_k<<<(SEEDN * CDIM + 255) / 256, 256, 0, stream>>>(srcF, tokS, S.memb, SEEDN, CDIM);
        xformer(stream, caa, srcF, srcF, S.memb, NF, SEEDN, S);

        // spot-guided sparse cross-attention -> new_ref / new_src (d_out slabs)
        sparse_layer(stream, sga, refF, srcF, spotR, newR, NF, NF, S);
        sparse_layer(stream, sga, srcF, refF, spotS, newS, NF, NF, S);
    }
}
